// SparseGlobalAttention_70042326663826
// MI455X (gfx1250) — compile-verified, matrix operand order checked by reading, not run on hardware
//
#include <hip/hip_runtime.h>

// ---------------------------------------------------------------------------
// Types for CDNA5 WMMA (gfx1250, wave32)
// ---------------------------------------------------------------------------
typedef __bf16 v16bf __attribute__((ext_vector_type(16)));
typedef float  v8f   __attribute__((ext_vector_type(8)));

union Frag {                // 16 bf16 = 32 bytes = two 16-byte chunks
    v16bf v;
    uint4 q[2];
};

__device__ __forceinline__ __bf16 f2bf(float f) {
    union { float f; unsigned u; } x;
    x.f = f;
    unsigned r = x.u + 0x7fffu + ((x.u >> 16) & 1u);   // round-to-nearest-even
    unsigned short h = (unsigned short)(r >> 16);
    __bf16 b;
    __builtin_memcpy(&b, &h, 2);
    return b;
}

__device__ __forceinline__ v8f wmma_bf16(const Frag& a, const Frag& b, v8f c) {
    return __builtin_amdgcn_wmma_f32_16x16x32_bf16(
        false, a.v, false, b.v, (short)0, c, false, false);
}

// Problem constants
#define BATCH 8
#define SEQ   8192
#define HID   1024
#define NH    16
#define HD    64
#define KSIG  1024
#define KV    1025   // KSIG + register row

// ---------------------------------------------------------------------------
// 1) Build idx[b][0..1023] = ascending positions where mask is true.
//    One wave (32 lanes) per batch; ballot-based stable compaction.
// ---------------------------------------------------------------------------
__global__ void build_idx_kernel(const unsigned char* __restrict__ mask,
                                 int* __restrict__ idx) {
    const int b = blockIdx.x;
    const int lane = threadIdx.x;   // 0..31
    int count = 0;
    for (int base = 0; base < SEQ; base += 32) {
        bool mv = mask[(size_t)b * SEQ + base + lane] != 0;
        unsigned bal = (unsigned)__ballot(mv);
        int pre = __popc(bal & ((1u << lane) - 1u));
        int pos = count + pre;
        if (mv && pos < KSIG) idx[b * KSIG + pos] = base + lane;
        count += __popc(bal);
    }
}

// ---------------------------------------------------------------------------
// 2) out = hidden_states (float4 grid-stride copy)
// ---------------------------------------------------------------------------
__global__ void copy_kernel(const float4* __restrict__ src,
                            float4* __restrict__ dst, long n4) {
    long i = (long)blockIdx.x * blockDim.x + threadIdx.x;
    long stride = (long)gridDim.x * blockDim.x;
    for (; i < n4; i += stride) dst[i] = src[i];
}

// ---------------------------------------------------------------------------
// 3) f32 -> bf16 conversion (weights)
// ---------------------------------------------------------------------------
__global__ void cvt_kernel(const float* __restrict__ src,
                           __bf16* __restrict__ dst, int n4) {
    int i = blockIdx.x * blockDim.x + threadIdx.x;
    int stride = gridDim.x * blockDim.x;
    for (; i < n4; i += stride) {
        float4 v = ((const float4*)src)[i];
        union { __bf16 o[4]; uint2 u; } u;
        u.o[0] = f2bf(v.x); u.o[1] = f2bf(v.y);
        u.o[2] = f2bf(v.z); u.o[3] = f2bf(v.w);
        ((uint2*)dst)[i] = u.u;
    }
}

// ---------------------------------------------------------------------------
// 4) signal_bf16[b][m][:] = bf16(hidden[b][idx[b][m]][:])
//    grid = (KSIG, BATCH), block = 256, 4 cols/thread
// ---------------------------------------------------------------------------
__global__ void gather_kernel(const float* __restrict__ hidden,
                              const int* __restrict__ idx,
                              __bf16* __restrict__ sig) {
    const int m = blockIdx.x;
    const int b = blockIdx.y;
    const int c = threadIdx.x * 4;
    const int row = idx[b * KSIG + m];
    float4 v = *(const float4*)(hidden + ((size_t)b * SEQ + row) * HID + c);
    union { __bf16 o[4]; uint2 u; } u;
    u.o[0] = f2bf(v.x); u.o[1] = f2bf(v.y);
    u.o[2] = f2bf(v.z); u.o[3] = f2bf(v.w);
    *(uint2*)(sig + ((size_t)b * KSIG + m) * HID + c) = u.u;
}

// ---------------------------------------------------------------------------
// 5) Register-row projection: row 1024 of the K (or V) buffer.
//    grid = BATCH, block = 256; each thread computes 4 outputs.
// ---------------------------------------------------------------------------
__global__ void regproj_kernel(const float* __restrict__ reg,
                               const float* __restrict__ W,
                               const float* __restrict__ bias,
                               __bf16* __restrict__ kv) {
    const int b = blockIdx.x;
    const int tid = threadIdx.x;
    __shared__ float sreg[HID];
    for (int i = tid; i < HID; i += 256) sreg[i] = reg[(size_t)b * HID + i];
    __syncthreads();
    for (int o = 0; o < 4; ++o) {
        int n = tid + o * 256;
        float acc = 0.f;
        const float4* wr = (const float4*)(W + (size_t)n * HID);
        #pragma unroll 4
        for (int k4 = 0; k4 < HID / 4; ++k4) {
            float4 w = wr[k4];
            acc += w.x * sreg[k4 * 4 + 0] + w.y * sreg[k4 * 4 + 1]
                 + w.z * sreg[k4 * 4 + 2] + w.w * sreg[k4 * 4 + 3];
        }
        kv[(size_t)b * KV * HID + (size_t)KSIG * HID + n] = f2bf(acc + bias[n]);
    }
}

// ---------------------------------------------------------------------------
// 6) Tiled WMMA GEMM:  C[m,n] = (sum_k A[m,k] * W[n,k] + bias[n]) * alpha
//    Block tile 64(M) x 128(N), K-step 32, DOUBLE-BUFFERED LDS (one barrier
//    per step; next tile's global loads overlap the WMMA chain). L2 prefetch
//    (global_prefetch_b8) issued two tiles ahead.
//    8 waves: wm=wave>>1 (4), wn=wave&1 (2), wave computes 16x64.
//    Output: bf16 buffer OR f32 buffer with optional row-scatter (O-proj).
//    grid = (N/128=8, M/64=16, BATCH), block = 256.
// ---------------------------------------------------------------------------
__global__ __launch_bounds__(256)
void gemm_wmma_kernel(const __bf16* __restrict__ A, long aStride,
                      const __bf16* __restrict__ W,
                      const float* __restrict__ bias, float alpha,
                      __bf16* __restrict__ outBf, float* __restrict__ outF,
                      long oStride, int ldo,
                      const int* __restrict__ scatter) {
    const int b = blockIdx.z;
    const int n0 = blockIdx.x * 128;
    const int m0 = blockIdx.y * 64;
    const int tid = threadIdx.x;
    const int wave = tid >> 5;
    const int lane = tid & 31;
    const int ln = lane & 15;
    const int half = lane >> 4;
    const int wm = wave >> 1;   // 0..3
    const int wn = wave & 1;    // 0..1

    __shared__ __align__(16) __bf16 lA[2][64 * 32];
    __shared__ __align__(16) __bf16 lB[2][128 * 32];

    const __bf16* Ab = A + (size_t)b * aStride;
    // per-thread staging coordinates
    const int ar = tid >> 2, aseg = (tid & 3) * 8;          // A: 8 bf16/thread
    const int br = tid >> 1, bsg = (tid & 1) * 16;          // B: 16 bf16/thread
    const __bf16* agp = Ab + (size_t)(m0 + ar) * HID + aseg;
    const __bf16* bgp = W + (size_t)(n0 + br) * HID + bsg;

    v8f acc[4] = {};

    // stage tile 0 into buffer 0
    *(uint4*)&lA[0][ar * 32 + aseg]    = *(const uint4*)(agp);
    *(uint4*)&lB[0][br * 32 + bsg]     = *(const uint4*)(bgp);
    *(uint4*)&lB[0][br * 32 + bsg + 8] = *(const uint4*)(bgp + 8);
    __syncthreads();

    const int NSTEP = HID / 32;   // 32
    for (int i = 0; i < NSTEP; ++i) {
        const int cur = i & 1, nxt = cur ^ 1;
        const bool hasNext = (i + 1) < NSTEP;

        // issue next tile's global loads early (overlap with WMMA below)
        uint4 ra, rb0, rb1;
        if (hasNext) {
            const int k1 = (i + 1) * 32;
            ra  = *(const uint4*)(agp + k1);
            rb0 = *(const uint4*)(bgp + k1);
            rb1 = *(const uint4*)(bgp + k1 + 8);
        }
        // prefetch two tiles ahead into cache hierarchy
        if (i + 2 < NSTEP) {
            const int k2 = (i + 2) * 32;
            __builtin_prefetch((const void*)(agp + k2), 0, 1);
            __builtin_prefetch((const void*)(bgp + k2), 0, 1);
        }

        // compute from current buffer
        Frag fa;
        const __bf16* ab = &lA[cur][(wm * 16 + ln) * 32 + 8 * half];
        fa.q[0] = *(const uint4*)(ab);
        fa.q[1] = *(const uint4*)(ab + 16);
        #pragma unroll
        for (int t = 0; t < 4; ++t) {
            Frag fb;
            const __bf16* bb = &lB[cur][(wn * 64 + t * 16 + ln) * 32 + 8 * half];
            fb.q[0] = *(const uint4*)(bb);
            fb.q[1] = *(const uint4*)(bb + 16);
            acc[t] = wmma_bf16(fa, fb, acc[t]);
        }

        // commit next tile to the other buffer
        if (hasNext) {
            *(uint4*)&lA[nxt][ar * 32 + aseg]    = ra;
            *(uint4*)&lB[nxt][br * 32 + bsg]     = rb0;
            *(uint4*)&lB[nxt][br * 32 + bsg + 8] = rb1;
        }
        __syncthreads();
    }

    // epilogue: C layout: lanes 0-15 -> M=r, lanes 16-31 -> M=r+8; N=ln
    #pragma unroll
    for (int t = 0; t < 4; ++t) {
        #pragma unroll
        for (int r = 0; r < 8; ++r) {
            int m = m0 + wm * 16 + r + 8 * half;
            int n = n0 + wn * 64 + t * 16 + ln;
            float v = (acc[t][r] + bias[n]) * alpha;
            int orow = scatter ? scatter[b * KSIG + m] : m;
            size_t off = (size_t)b * oStride + (size_t)orow * ldo + n;
            if (outBf) outBf[off] = f2bf(v);
            else       outF[off] = v;
        }
    }
}

// ---------------------------------------------------------------------------
// 7) Attention: per (qchunk, head, batch). Block = 256 = 8 waves; each wave
//    owns 16 query rows, loops over 33 chunks of 32 keys (1025 total) with
//    online softmax. K tile + transposed-V tile DOUBLE-BUFFERED in LDS so
//    the next chunk's global loads overlap this chunk's 8 WMMAs + softmax.
//    P is re-laid-out (C-frag -> A-frag) via wave-private LDS.
//    grid = (8, NH, BATCH)
// ---------------------------------------------------------------------------
__global__ __launch_bounds__(256)
void attn_kernel(const __bf16* __restrict__ Qb, const __bf16* __restrict__ Kb,
                 const __bf16* __restrict__ Vb, __bf16* __restrict__ AO) {
    const int b = blockIdx.z;
    const int h = blockIdx.y;
    const int qc = blockIdx.x;
    const int tid = threadIdx.x;
    const int wave = tid >> 5;
    const int lane = tid & 31;
    const int ln = lane & 15;
    const int half = lane >> 4;

    __shared__ __align__(16) __bf16 lK[2][32 * 64];    // [j][d]
    __shared__ __align__(16) __bf16 lVt[2][64 * 32];   // [d][j]
    __shared__ __align__(16) __bf16 lP[8][16 * 32];    // per-wave [m][j]

    const size_t qBatch = (size_t)KSIG * HID;
    const size_t kvBatch = (size_t)KV * HID;
    const __bf16* Qp = Qb + (size_t)b * qBatch + (size_t)h * HD;
    const __bf16* Kp = Kb + (size_t)b * kvBatch + (size_t)h * HD;
    const __bf16* Vp = Vb + (size_t)b * kvBatch + (size_t)h * HD;

    const int q0 = qc * 128 + wave * 16;
    const int qrow = q0 + ln;

    // preload Q fragments (2 K-chunks over d=64). Softmax scale 1/sqrt(HD)
    // was folded into Q at the Q-GEMM epilogue.
    Frag fq[2];
    #pragma unroll
    for (int kc = 0; kc < 2; ++kc) {
        const __bf16* base = Qp + (size_t)qrow * HID + kc * 32 + 8 * half;
        fq[kc].q[0] = *(const uint4*)(base);
        fq[kc].q[1] = *(const uint4*)(base + 16);
    }

    v8f acc[4] = {};
    float mrow[8], lrow[8];
    #pragma unroll
    for (int r = 0; r < 8; ++r) { mrow[r] = -3.0e38f; lrow[r] = 0.f; }

    // staging coordinates: 8 bf16 per thread per matrix
    const int sj = tid >> 3;            // 0..31
    const int sseg = (tid & 7) * 8;     // 0..56

    // stage chunk 0 into buffer 0 (chunk 0 is fully in range)
    {
        uint4 kv4 = *(const uint4*)(Kp + (size_t)sj * HID + sseg);
        uint4 vv4 = *(const uint4*)(Vp + (size_t)sj * HID + sseg);
        *(uint4*)&lK[0][sj * 64 + sseg] = kv4;
        __bf16 tmp[8];
        __builtin_memcpy(tmp, &vv4, 16);
        #pragma unroll
        for (int e = 0; e < 8; ++e) lVt[0][(sseg + e) * 32 + sj] = tmp[e];
    }
    __syncthreads();

    const int NCHUNK = 33;              // ceil(1025 / 32)
    for (int jc = 0; jc < NCHUNK; ++jc) {
        const int j0 = jc * 32;
        const int cur = jc & 1, nxt = cur ^ 1;
        const bool hasNext = (jc + 1) < NCHUNK;

        // issue next chunk's K/V global loads early (zero-pad OOB rows)
        uint4 kv4 = make_uint4(0, 0, 0, 0);
        uint4 vv4 = make_uint4(0, 0, 0, 0);
        if (hasNext) {
            const int jn = (jc + 1) * 32 + sj;
            if (jn < KV) {
                kv4 = *(const uint4*)(Kp + (size_t)jn * HID + sseg);
                vv4 = *(const uint4*)(Vp + (size_t)jn * HID + sseg);
            }
        }

        // scores: S = Q K^T (two 16-col halves, d accumulated over 2 chunks)
        v8f s0 = {}, s1 = {};
        #pragma unroll
        for (int kc = 0; kc < 2; ++kc) {
            Frag fk0, fk1;
            const __bf16* k0p = &lK[cur][ln * 64 + kc * 32 + 8 * half];
            fk0.q[0] = *(const uint4*)(k0p);
            fk0.q[1] = *(const uint4*)(k0p + 16);
            const __bf16* k1p = &lK[cur][(ln + 16) * 64 + kc * 32 + 8 * half];
            fk1.q[0] = *(const uint4*)(k1p);
            fk1.q[1] = *(const uint4*)(k1p + 16);
            s0 = wmma_bf16(fq[kc], fk0, s0);
            s1 = wmma_bf16(fq[kc], fk1, s1);
        }

        const bool v0 = (j0 + ln) < KV;
        const bool v1 = (j0 + 16 + ln) < KV;

        // online softmax update (per-row stats via 16-lane reductions)
        #pragma unroll
        for (int r = 0; r < 8; ++r) {
            float a0 = v0 ? s0[r] : -3.0e38f;
            float a1 = v1 ? s1[r] : -3.0e38f;
            float cm = fmaxf(a0, a1);
            cm = fmaxf(cm, __shfl_xor(cm, 1));
            cm = fmaxf(cm, __shfl_xor(cm, 2));
            cm = fmaxf(cm, __shfl_xor(cm, 4));
            cm = fmaxf(cm, __shfl_xor(cm, 8));
            float mn = fmaxf(mrow[r], cm);
            float p0 = __expf(a0 - mn);
            float p1 = __expf(a1 - mn);
            float rs = p0 + p1;
            rs += __shfl_xor(rs, 1);
            rs += __shfl_xor(rs, 2);
            rs += __shfl_xor(rs, 4);
            rs += __shfl_xor(rs, 8);
            float fr = __expf(mrow[r] - mn);
            lrow[r] = lrow[r] * fr + rs;
            mrow[r] = mn;
            acc[0][r] *= fr; acc[1][r] *= fr;
            acc[2][r] *= fr; acc[3][r] *= fr;
            int m = r + 8 * half;
            lP[wave][m * 32 + ln]      = f2bf(p0);
            lP[wave][m * 32 + 16 + ln] = f2bf(p1);
        }

        // PV: acc[t] += P (16x32) x V (32x16)
        Frag fp;
        const __bf16* pb = &lP[wave][ln * 32 + 8 * half];
        fp.q[0] = *(const uint4*)(pb);
        fp.q[1] = *(const uint4*)(pb + 16);
        #pragma unroll
        for (int t = 0; t < 4; ++t) {
            Frag fv;
            const __bf16* vb = &lVt[cur][(t * 16 + ln) * 32 + 8 * half];
            fv.q[0] = *(const uint4*)(vb);
            fv.q[1] = *(const uint4*)(vb + 16);
            acc[t] = wmma_bf16(fp, fv, acc[t]);
        }

        // commit next chunk into the other buffer
        if (hasNext) {
            *(uint4*)&lK[nxt][sj * 64 + sseg] = kv4;
            __bf16 tmp[8];
            __builtin_memcpy(tmp, &vv4, 16);
            #pragma unroll
            for (int e = 0; e < 8; ++e) lVt[nxt][(sseg + e) * 32 + sj] = tmp[e];
        }
        __syncthreads();
    }

    // normalize + store AO (bf16) interleaved as (B, KSIG, NH*HD)
    #pragma unroll
    for (int t = 0; t < 4; ++t) {
        #pragma unroll
        for (int r = 0; r < 8; ++r) {
            int m = r + 8 * half;
            float val = acc[t][r] / lrow[r];
            AO[(size_t)b * qBatch + (size_t)(q0 + m) * HID + h * HD + t * 16 + ln]
                = f2bf(val);
        }
    }
}

// ---------------------------------------------------------------------------
// Host-side orchestration
// ---------------------------------------------------------------------------
extern "C" void kernel_launch(void* const* d_in, const int* in_sizes, int n_in,
                              void* d_out, int out_size, void* d_ws, size_t ws_size,
                              hipStream_t stream) {
    const float* hidden = (const float*)d_in[0];
    const unsigned char* mask = (const unsigned char*)d_in[1];
    const float* reg = (const float*)d_in[2];
    const float* Wq = (const float*)d_in[3];  const float* bq = (const float*)d_in[4];
    const float* Wk = (const float*)d_in[5];  const float* bk = (const float*)d_in[6];
    const float* Wv = (const float*)d_in[7];  const float* bv = (const float*)d_in[8];
    const float* Wo = (const float*)d_in[9];  const float* bo = (const float*)d_in[10];
    const float* Wrk = (const float*)d_in[11]; const float* brk = (const float*)d_in[12];
    const float* Wrv = (const float*)d_in[13]; const float* brv = (const float*)d_in[14];
    float* out = (float*)d_out;

    // workspace layout (256-byte aligned slabs)
    char* ws = (char*)d_ws;
    size_t off = 0;
    auto alloc = [&](size_t bytes) {
        size_t p = off; off += (bytes + 255) & ~(size_t)255; return p;
    };
    const size_t szIdx = (size_t)BATCH * KSIG * sizeof(int);
    const size_t szSig = (size_t)BATCH * KSIG * HID * sizeof(__bf16);
    const size_t szKV  = (size_t)BATCH * KV * HID * sizeof(__bf16);
    const size_t szW   = (size_t)HID * HID * sizeof(__bf16);

    int* idx       = (int*)(ws + alloc(szIdx));
    __bf16* sig    = (__bf16*)(ws + alloc(szSig));
    __bf16* Qbuf   = (__bf16*)(ws + alloc(szSig));
    __bf16* Kbuf   = (__bf16*)(ws + alloc(szKV));
    __bf16* Vbuf   = (__bf16*)(ws + alloc(szKV));
    __bf16* AObuf  = (__bf16*)(ws + alloc(szSig));
    __bf16* Wq_b   = (__bf16*)(ws + alloc(szW));
    __bf16* Wk_b   = (__bf16*)(ws + alloc(szW));
    __bf16* Wv_b   = (__bf16*)(ws + alloc(szW));
    __bf16* Wo_b   = (__bf16*)(ws + alloc(szW));
    (void)ws_size; (void)in_sizes; (void)n_in; (void)out_size;

    // 1) index compaction
    build_idx_kernel<<<BATCH, 32, 0, stream>>>(mask, idx);

    // 2) out = hidden (float4)
    {
        long n4 = (long)BATCH * SEQ * HID / 4;
        copy_kernel<<<4096, 256, 0, stream>>>((const float4*)hidden,
                                              (float4*)out, n4);
    }

    // 3) weight conversion
    {
        int n4 = HID * HID / 4;
        cvt_kernel<<<1024, 256, 0, stream>>>(Wq, Wq_b, n4);
        cvt_kernel<<<1024, 256, 0, stream>>>(Wk, Wk_b, n4);
        cvt_kernel<<<1024, 256, 0, stream>>>(Wv, Wv_b, n4);
        cvt_kernel<<<1024, 256, 0, stream>>>(Wo, Wo_b, n4);
    }

    // 4) gather signal rows (bf16)
    gather_kernel<<<dim3(KSIG, BATCH), 256, 0, stream>>>(hidden, idx, sig);

    // 5) register K/V rows
    regproj_kernel<<<BATCH, 256, 0, stream>>>(reg, Wrk, brk, Kbuf);
    regproj_kernel<<<BATCH, 256, 0, stream>>>(reg, Wrv, brv, Vbuf);

    // 6) Q/K/V GEMMs (WMMA bf16). Q has softmax scale 1/sqrt(64)=0.125 folded in.
    const dim3 ggrid(HID / 128, KSIG / 64, BATCH);
    gemm_wmma_kernel<<<ggrid, 256, 0, stream>>>(
        sig, (long)KSIG * HID, Wq_b, bq, 0.125f,
        Qbuf, nullptr, (long)KSIG * HID, HID, nullptr);
    gemm_wmma_kernel<<<ggrid, 256, 0, stream>>>(
        sig, (long)KSIG * HID, Wk_b, bk, 1.0f,
        Kbuf, nullptr, (long)KV * HID, HID, nullptr);
    gemm_wmma_kernel<<<ggrid, 256, 0, stream>>>(
        sig, (long)KSIG * HID, Wv_b, bv, 1.0f,
        Vbuf, nullptr, (long)KV * HID, HID, nullptr);

    // 7) attention
    attn_kernel<<<dim3(KSIG / 128, NH, BATCH), 256, 0, stream>>>(
        Qbuf, Kbuf, Vbuf, AObuf);

    // 8) O-projection, scattered into out at idx rows (f32 output)
    gemm_wmma_kernel<<<ggrid, 256, 0, stream>>>(
        AObuf, (long)KSIG * HID, Wo_b, bo, 1.0f,
        nullptr, out, (long)SEQ * HID, HID, idx);
}
